// DarkNet_7327214207629
// MI455X (gfx1250) — matrix operand order repaired and checked by reading, not compiled, not run on hardware
//
#include <hip/hip_runtime.h>
#include <hip/hip_bf16.h>

// ---------------------------------------------------------------------------
// Types for CDNA5 WMMA
// ---------------------------------------------------------------------------
typedef __attribute__((ext_vector_type(16))) __bf16      v16bf;
typedef __attribute__((ext_vector_type(8)))  float       v8f;
typedef __attribute__((ext_vector_type(8)))  unsigned    v8u;
typedef __attribute__((ext_vector_type(4)))  float       v4f;

__device__ __forceinline__ unsigned short f2bf(float f) {
    unsigned u = __float_as_uint(f);
    unsigned r = u + 0x7FFFu + ((u >> 16) & 1u);   // round-to-nearest-even
    return (unsigned short)(r >> 16);
}

// ---------------------------------------------------------------------------
// Fold BN into scale/bias: y = scale*conv + bias
// ---------------------------------------------------------------------------
__global__ void bn_fold_kernel(const float* __restrict__ ga, const float* __restrict__ be,
                               const float* __restrict__ me, const float* __restrict__ va,
                               float* __restrict__ sc, float* __restrict__ bi, int C) {
    int i = blockIdx.x * blockDim.x + threadIdx.x;
    if (i < C) {
        float s = ga[i] * rsqrtf(va[i] + 1e-5f);
        sc[i] = s;
        bi[i] = be[i] - me[i] * s;
    }
}

// ---------------------------------------------------------------------------
// Implicit-GEMM conv + BN + leaky (+residual), bf16 WMMA, f32 accumulate
// M = Cout, N = B*Ho*Wo, K = Cin*KH*KW
// Block tile 64x256x32, 8 waves; wave tile 32x64 = 2x4 WMMA accumulators.
// Templated on (KH*KW, KW) so im2col decode is constant mul/shift.
// ---------------------------------------------------------------------------
#define BM 64
#define BN_ 256
#define BK 32

struct ConvP {
    const float* X; const float* W; const float* scale; const float* bias;
    const float* resid; float* Y;
    int Bn, Cin, H, Wd, Cout, Ho, Wo, stride, pad;
    int M, N, K;
};

template <int KHW_, int KW_>
__global__ __launch_bounds__(256) void conv_wmma_kernel(ConvP p) {
    __shared__ __align__(16) unsigned short Alds[BM * BK];   // [row][k]
    __shared__ __align__(16) unsigned short Blds[BN_ * BK];  // [col][k]

    const int tid  = threadIdx.x;
    const int lane = tid & 31;
    const int wave = tid >> 5;
    const int wm   = wave & 1;      // 2 row bands of 32
    const int wn   = wave >> 1;     // 4 col bands of 64
    const int tileM = blockIdx.y * BM;
    const int tileN = blockIdx.x * BN_;
    const int HW   = p.Ho * p.Wo;

    // --- B staging: each thread owns one n-column, all 32 k values ---
    const int ncolB = tid;                 // 0..255
    int nB = tileN + ncolB;
    bool nBv = nB < p.N;
    int bB = 0, ohB = 0, owB = 0;
    if (nBv) {
        bB = nB / HW; int r = nB - bB * HW;
        ohB = r / p.Wo; owB = r - ohB * p.Wo;
    }
    const int ihB0 = ohB * p.stride - p.pad;
    const int iwB0 = owB * p.stride - p.pad;
    // --- A staging: thread -> row tid>>2, k-octet (tid&3)*8 ---
    const int arow = tid >> 2;
    const int akb  = (tid & 3) * 8;

    v8f acc[2][4] = {};

    const int ksteps = (p.K + BK - 1) / BK;
    for (int kt = 0; kt < ksteps; ++kt) {
        const int k0 = kt * BK;

        // ---- stage A tile (weights are row-major [M][K] already) ----
        {
            int gm = tileM + arow;
            const float* wp = p.W + (size_t)gm * p.K + (k0 + akb);
            if (kt + 1 < ksteps)
                __builtin_prefetch(wp + BK, 0, 1);   // global_prefetch next k-tile
            if (gm < p.M && (k0 + akb + 8) <= p.K) { // fast interior: 2x b128 loads
                v4f w0 = *(const v4f*)(wp);
                v4f w1 = *(const v4f*)(wp + 4);
                unsigned short* d = &Alds[arow * BK + akb];
                #pragma unroll
                for (int e = 0; e < 4; ++e) { d[e] = f2bf(w0[e]); d[4 + e] = f2bf(w1[e]); }
            } else {
                #pragma unroll
                for (int e = 0; e < 8; ++e) {
                    int gk = k0 + akb + e;
                    float w = (gm < p.M && gk < p.K) ? wp[e] : 0.0f;
                    Alds[arow * BK + akb + e] = f2bf(w);
                }
            }
        }
        // ---- stage B tile (im2col gather), col-major so k-pairs contiguous ----
        {
            #pragma unroll
            for (int e = 0; e < BK; ++e) {
                int gk = k0 + e;
                float xv = 0.0f;
                if (nBv && gk < p.K) {
                    int ci = gk / KHW_;                // constant divisor -> mul/shift
                    int rem = gk - ci * KHW_;
                    int kh = rem / KW_;
                    int kw = rem - kh * KW_;
                    int ih = ihB0 + kh;
                    int iw = iwB0 + kw;
                    if (ih >= 0 && ih < p.H && iw >= 0 && iw < p.Wd)
                        xv = p.X[(((size_t)bB * p.Cin + ci) * p.H + ih) * p.Wd + iw];
                }
                Blds[ncolB * BK + e] = f2bf(xv);
            }
        }
        __syncthreads();

        // ---- assemble fragments per documented CDNA5 16-bit layouts ----
        const int lrow = lane & 15;
        const int sel  = lane >> 4;
        v8u au[2], bu[4];
        #pragma unroll
        for (int pi = 0; pi < 8; ++pi) {
            // A: lane<16 -> K 0..7 / 16..23 ; lane>=16 -> +8
            int ka = sel * 8 + (pi < 4 ? 2 * pi : 16 + 2 * (pi - 4));
            au[0][pi] = *(const unsigned*)&Alds[(wm * 32 + lrow     ) * BK + ka];
            au[1][pi] = *(const unsigned*)&Alds[(wm * 32 + 16 + lrow) * BK + ka];
            // B: lane<16 -> K 0..7 / 16..23 ; lane>=16 -> +8
            int kb = sel * 8 + (pi < 4 ? 2 * pi : 16 + 2 * (pi - 4));
            #pragma unroll
            for (int j = 0; j < 4; ++j)
                bu[j][pi] = *(const unsigned*)&Blds[(wn * 64 + j * 16 + lrow) * BK + kb];
        }
        v16bf af[2], bfr[4];
        af[0] = __builtin_bit_cast(v16bf, au[0]);
        af[1] = __builtin_bit_cast(v16bf, au[1]);
        #pragma unroll
        for (int j = 0; j < 4; ++j) bfr[j] = __builtin_bit_cast(v16bf, bu[j]);

        #pragma unroll
        for (int i = 0; i < 2; ++i)
            #pragma unroll
            for (int j = 0; j < 4; ++j)
                acc[i][j] = __builtin_amdgcn_wmma_f32_16x16x32_bf16(
                    false, af[i], false, bfr[j], (short)0, acc[i][j], false, false);

        __syncthreads();
    }

    // ---- epilogue: BN + leaky(0.1) + optional residual, NCHW scatter ----
    const int lcol = lane & 15;
    const int msel = (lane >> 4) * 8;
    #pragma unroll
    for (int j = 0; j < 4; ++j) {
        int n = tileN + wn * 64 + j * 16 + lcol;
        if (n >= p.N) continue;
        int b = n / HW; int r0 = n - b * HW;
        int oh = r0 / p.Wo; int ow = r0 - oh * p.Wo;
        #pragma unroll
        for (int i = 0; i < 2; ++i) {
            int mbase = tileM + wm * 32 + i * 16 + msel;
            #pragma unroll
            for (int r = 0; r < 8; ++r) {
                int m = mbase + r;
                if (m < p.M) {
                    float v = acc[i][j][r] * p.scale[m] + p.bias[m];
                    v = v >= 0.0f ? v : 0.1f * v;
                    size_t off = (((size_t)b * p.Cout + m) * p.Ho + oh) * p.Wo + ow;
                    if (p.resid) v += p.resid[off];
                    p.Y[off] = v;
                }
            }
        }
    }
}

// ---------------------------------------------------------------------------
// Attention helper kernels (small, VALU)
// ---------------------------------------------------------------------------
__global__ void pool_kernel(const float* __restrict__ X, float* __restrict__ pmax,
                            float* __restrict__ pavg, int HW) {
    __shared__ float smx[256], ssm[256];
    int bc = blockIdx.x, t = threadIdx.x;
    const float* p = X + (size_t)bc * HW;
    float mx = -3.4e38f, sm = 0.f;
    for (int i = t; i < HW; i += 256) { float v = p[i]; mx = fmaxf(mx, v); sm += v; }
    smx[t] = mx; ssm[t] = sm; __syncthreads();
    for (int s = 128; s > 0; s >>= 1) {
        if (t < s) { smx[t] = fmaxf(smx[t], smx[t + s]); ssm[t] += ssm[t + s]; }
        __syncthreads();
    }
    if (t == 0) { pmax[bc] = smx[0]; pavg[bc] = ssm[0] / (float)HW; }
}

// vin/vout layout: [2][B][C]  (z=0 max-path, z=1 avg-path)
__global__ void gemv_act_kernel(const float* __restrict__ Wm, const float* __restrict__ vin,
                                float* __restrict__ vout, int C, int Bn, int act) {
    int j = blockIdx.x * blockDim.x + threadIdx.x;
    int b = blockIdx.y, z = blockIdx.z;
    if (j >= C) return;
    const float* v  = vin + ((size_t)z * Bn + b) * C;
    const float* wr = Wm + (size_t)j * C;
    float s = 0.f;
    for (int i = 0; i < C; ++i) s += wr[i] * v[i];
    vout[((size_t)z * Bn + b) * C + j] = act ? 1.f / (1.f + __expf(-s)) : fmaxf(s, 0.f);
}

__global__ void ca_apply_kernel(const float* __restrict__ X, const float* __restrict__ g,
                                float* __restrict__ Y, int Bn, int C, int HW) {
    size_t idx = (size_t)blockIdx.x * blockDim.x + threadIdx.x;
    if (idx >= (size_t)Bn * C * HW) return;
    int bc = (int)(idx / HW);
    int b = bc / C, c = bc - b * C;
    float m = g[(size_t)b * C + c];
    float a = g[((size_t)Bn + b) * C + c];
    float w_hi = fminf(m, a);
    float w_lo = fminf(1.f - m, 1.f - a);
    float w_md = fmaxf(fminf(m, 1.f - a), fminf(1.f - m, a));
    float ca = (w_hi + 0.5f * w_md) / (w_hi + w_md + w_lo + 1e-6f);
    Y[idx] = ca * X[idx];
}

__global__ void smap_kernel(const float* __restrict__ X, float* __restrict__ S,
                            int Bn, int C, int HW) {
    int idx = blockIdx.x * blockDim.x + threadIdx.x;
    if (idx >= Bn * HW) return;
    int b = idx / HW, hw = idx - b * HW;
    const float* p = X + (size_t)b * C * HW + hw;
    float mx = -3.4e38f, sm = 0.f;
    for (int c = 0; c < C; ++c) { float v = p[(size_t)c * HW]; mx = fmaxf(mx, v); sm += v; }
    S[((size_t)b * 2 + 0) * HW + hw] = mx;
    S[((size_t)b * 2 + 1) * HW + hw] = sm / (float)C;
}

__global__ void spatial_apply_kernel(const float* __restrict__ S, const float* __restrict__ cho,
                                     float* __restrict__ Y,
                                     const float* __restrict__ w3, const float* __restrict__ w5,
                                     const float* __restrict__ w7, int Bn, int C, int H, int W) {
    int idx = blockIdx.x * blockDim.x + threadIdx.x;
    if (idx >= Bn * H * W) return;
    int b = idx / (H * W); int hw = idx - b * (H * W);
    int h = hw / W; int w = hw - h * W;
    const float* sb = S + (size_t)b * 2 * H * W;
    const float* wt[3] = { w3, w5, w7 };
    float am[3];
    #pragma unroll
    for (int d1 = 0; d1 < 3; ++d1) {
        int d = d1 + 1;
        float acc = 0.f;
        for (int ch = 0; ch < 2; ++ch)
            for (int kh = 0; kh < 3; ++kh)
                for (int kw = 0; kw < 3; ++kw) {
                    int ih = h - d + kh * d, iw = w - d + kw * d;
                    if (ih >= 0 && ih < H && iw >= 0 && iw < W)
                        acc += wt[d1][(ch * 3 + kh) * 3 + kw] * sb[(size_t)ch * H * W + ih * W + iw];
                }
        am[d1] = 1.f / (1.f + __expf(-acc));
    }
    float w_hi = fminf(fminf(am[0], am[1]), am[2]);
    float w_lo = fminf(fminf(1.f - am[0], 1.f - am[1]), 1.f - am[2]);
    float w_md = fminf(fmaxf(1.f - w_hi - w_lo, 0.f), 1.f);
    float sa = (w_hi + 0.5f * w_md) / (w_hi + w_md + w_lo + 1e-6f);
    size_t base = (size_t)b * C * H * W + hw;
    for (int c = 0; c < C; ++c) Y[base + (size_t)c * H * W] = sa * cho[base + (size_t)c * H * W];
}

// ---------------------------------------------------------------------------
// Host orchestration
// ---------------------------------------------------------------------------
extern "C" void kernel_launch(void* const* d_in, const int* in_sizes, int n_in,
                              void* d_out, int out_size, void* d_ws, size_t ws_size,
                              hipStream_t stream) {
    (void)out_size; (void)ws_size;
    const int XSZ = 4 * 3 * 416 * 416;
    int xi = 0;
    for (int i = 0; i < n_in; ++i) if (in_sizes[i] == XSZ) { xi = i; break; }
    const float* x_in = (const float*)d_in[xi];

    int cur_i = 0;
    auto nx = [&]() -> const float* {
        if (cur_i == xi) ++cur_i;
        int i = (cur_i < n_in) ? cur_i : (n_in - 1);
        ++cur_i;
        return (const float*)d_in[i];
    };
    struct BNp { const float *be, *ga, *me, *va; };
    auto nbn = [&]() { BNp r; r.be = nx(); r.ga = nx(); r.me = nx(); r.va = nx(); return r; };

    // sorted-key pytree order: bn1, conv1, layers, mlps, spatial
    BNp bn1 = nbn();
    const float* conv1w = nx();

    const int NBLK[5] = { 1, 2, 8, 8, 4 };
    const int P0[5] = { 32, 64, 128, 256, 512 };
    const int P1[5] = { 64, 128, 256, 512, 1024 };
    struct Blk { BNp b1, b2; const float *c1, *c2; };
    struct Stg { Blk blk[8]; BNp dsbn; const float* dsw; };
    Stg st[5];
    for (int i = 0; i < 5; ++i) {
        for (int j = 0; j < NBLK[i]; ++j) {              // 'blocks' < 'ds_bn' < 'ds_w'
            st[i].blk[j].b1 = nbn();                     // bn1 < bn2 < c1 < c2
            st[i].blk[j].b2 = nbn();
            st[i].blk[j].c1 = nx();
            st[i].blk[j].c2 = nx();
        }
        st[i].dsbn = nbn();
        st[i].dsw  = nx();
    }
    const float *mw1[5], *mw2[5];
    for (int i = 0; i < 5; ++i) { mw1[i] = nx(); mw2[i] = nx(); }
    const float* sc3 = nx(); const float* sc5 = nx(); const float* sc7 = nx();

    // workspace layout (floats), ~180 MB total
    float* ws   = (float*)d_ws;
    float* bufA = ws;                       // 22,151,168 (4x32x416x416)
    float* bufB = bufA + 22151168;          // 11,075,584
    float* bufC = bufB + 11075584;          // 11,075,584
    float* sbuf = bufC + 11075584;          // 1024 scale
    float* bbuf = sbuf + 1024;              // 1024 bias
    float* pool = bbuf + 1024;              // [2][B][C<=1024] = 8192
    float* hbuf = pool + 8192;              // 8192
    float* gbuf = hbuf + 8192;              // 8192
    float* smap = gbuf + 8192;              // <= 346,112

    const int Bn = 4;

    auto run_conv = [&](const float* X, const float* Wt, BNp bn, const float* resid,
                        float* Y, int Cin, int H, int Cout, int KH, int stride, int pad) {
        bn_fold_kernel<<<(Cout + 255) / 256, 256, 0, stream>>>(bn.ga, bn.be, bn.me, bn.va,
                                                               sbuf, bbuf, Cout);
        ConvP p{};
        p.X = X; p.W = Wt; p.scale = sbuf; p.bias = bbuf; p.resid = resid; p.Y = Y;
        p.Bn = Bn; p.Cin = Cin; p.H = H; p.Wd = H; p.Cout = Cout;
        p.Ho = (H + 2 * pad - KH) / stride + 1; p.Wo = p.Ho;
        p.stride = stride; p.pad = pad;
        p.M = Cout; p.N = Bn * p.Ho * p.Wo; p.K = Cin * KH * KH;
        dim3 grid((p.N + BN_ - 1) / BN_, (p.M + BM - 1) / BM);
        if (KH == 1) conv_wmma_kernel<1, 1><<<grid, 256, 0, stream>>>(p);
        else         conv_wmma_kernel<9, 3><<<grid, 256, 0, stream>>>(p);
    };

    auto run_attn = [&](const float* X, float* cho, float* out, int C, int H,
                        const float* w1, const float* w2) {
        int HW = H * H;
        pool_kernel<<<Bn * C, 256, 0, stream>>>(X, pool, pool + Bn * C, HW);
        dim3 gg((C + 255) / 256, Bn, 2);
        gemv_act_kernel<<<gg, 256, 0, stream>>>(w1, pool, hbuf, C, Bn, 0);
        gemv_act_kernel<<<gg, 256, 0, stream>>>(w2, hbuf, gbuf, C, Bn, 1);
        size_t tot = (size_t)Bn * C * HW;
        ca_apply_kernel<<<(unsigned)((tot + 255) / 256), 256, 0, stream>>>(X, gbuf, cho, Bn, C, HW);
        smap_kernel<<<(Bn * HW + 255) / 256, 256, 0, stream>>>(cho, smap, Bn, C, HW);
        spatial_apply_kernel<<<(Bn * HW + 255) / 256, 256, 0, stream>>>(smap, cho, out,
                                                                        sc3, sc5, sc7, Bn, C, H, H);
    };

    // ---- forward ----
    run_conv(x_in, conv1w, bn1, nullptr, bufA, 3, 416, 32, 3, 1, 1);
    float* cur = bufA; float* fa = bufB; float* fb = bufC;
    int H = 416, Cc = 32;
    size_t out_off = 0;
    float* outf = (float*)d_out;

    for (int i = 0; i < 5; ++i) {
        int Cout = P1[i];
        run_conv(cur, st[i].dsw, st[i].dsbn, nullptr, fa, Cc, H, Cout, 3, 2, 1);
        { float* t = cur; cur = fa; fa = t; }
        H /= 2; Cc = Cout;
        for (int j = 0; j < NBLK[i]; ++j) {
            run_conv(cur, st[i].blk[j].c1, st[i].blk[j].b1, nullptr, fa, Cout, H, P0[i], 1, 1, 0);
            run_conv(fa,  st[i].blk[j].c2, st[i].blk[j].b2, cur,     fb, P0[i], H, Cout, 3, 1, 1);
            { float* t = cur; cur = fb; fb = t; }
        }
        run_attn(cur, fa, fb, Cout, H, mw1[i], mw2[i]);
        { float* t = cur; cur = fb; fb = t; }
        if (i >= 2) {
            size_t nel = (size_t)Bn * Cout * H * H;
            hipMemcpyAsync(outf + out_off, cur, nel * sizeof(float),
                           hipMemcpyDeviceToDevice, stream);
            out_off += nel;
        }
    }
}
